// GNNModel_45707041964386
// MI455X (gfx1250) — compile-verified, matrix operand order checked by reading, not run on hardware
//
#include <hip/hip_runtime.h>
#include <hip/hip_bf16.h>
#include <stdint.h>

// Problem constants (fixed by reference)
#define NN  100000
#define EEg 300000
#define EAg 400000
#define GGr 2048
#define HIDd 256
#define NH  4
#define HC  64

typedef __attribute__((ext_vector_type(16))) __bf16 v16bf;
typedef __attribute__((ext_vector_type(8)))  __bf16 v8bf;
typedef __attribute__((ext_vector_type(8)))  float  v8f;

// ---------------- helpers ----------------
__device__ __forceinline__ int enc_f(float f) {
  int s = __float_as_int(f);
  return s < 0 ? (s ^ 0x7fffffff) : s;          // order-preserving float->int
}
__device__ __forceinline__ float dec_f(int s) {
  return __int_as_float(s < 0 ? (s ^ 0x7fffffff) : s);
}

// ---------------- WMMA bf16 GEMM: C[M,Nout] = A[M,K] * Wt^T (+bias) ----------------
// A  : bf16 [M, K] row-major (activations pre-converted once per layer)
// Wt : bf16 [Nout, K] row-major (weights pre-transposed+converted)
// Block = 128 threads = 4 waves; each wave -> 16 rows x 64 cols (4 N-tiles).
// The block's B tile (64 x K bf16, <= 64KB) is staged in LDS once, so the
// K-loop is pure ds_load_b128 + global_load_b128(A) + v_wmma.
__global__ __launch_bounds__(128)
void k_gemm(const __bf16* __restrict__ A, const __bf16* __restrict__ Wt,
            const float* __restrict__ bias, float* __restrict__ C,
            int M, int K, int Nout)
{
  __shared__ __bf16 lds_b[64 * 512];          // K <= 512

  const int lane = threadIdx.x & 31;
  const int wave = threadIdx.x >> 5;
  const int l15  = lane & 15;
  const int half = lane >> 4;                 // 0: K 0..7/16..23 ; 1: K 8..15/24..31
  const int m0   = (blockIdx.x * 4 + wave) * 16;
  const int n0   = blockIdx.y * 64;
  const int row  = m0 + l15;
  const int rowc = row < M ? row : M - 1;     // clamp loads; keep EXEC all-ones for WMMA
  const int koff = half * 8;

  // ---- stage B tile: Wt rows n0..n0+63 are contiguous = exactly the tile ----
  {
    const uint4* bsrc = (const uint4*)(Wt + (size_t)n0 * K);
    uint4* bdst = (uint4*)lds_b;
    const int total16 = (64 * K * 2) / 16;    // # of 16B chunks (multiple of 128)
    for (int i = threadIdx.x; i < total16; i += 128) bdst[i] = bsrc[i];
  }
  __syncthreads();

  const __bf16* arow = A + (size_t)rowc * K;

  v8f acc[4] = {v8f{}, v8f{}, v8f{}, v8f{}};
#pragma unroll 2
  for (int k0 = 0; k0 < K; k0 += 32) {
    const __bf16* ap = arow + (k0 + koff);
    __builtin_prefetch(ap + 32, 0, 1);        // next K-chunk -> global_prefetch_b8
    v8bf alo = *(const v8bf*)(ap);
    v8bf ahi = *(const v8bf*)(ap + 16);
    v16bf a = __builtin_shufflevector(alo, ahi, 0, 1, 2, 3, 4, 5, 6, 7,
                                      8, 9, 10, 11, 12, 13, 14, 15);
#pragma unroll
    for (int j = 0; j < 4; ++j) {
      const __bf16* bp = lds_b + (size_t)(j * 16 + l15) * K + (k0 + koff);
      v8bf blo = *(const v8bf*)(bp);
      v8bf bhi = *(const v8bf*)(bp + 16);
      v16bf b = __builtin_shufflevector(blo, bhi, 0, 1, 2, 3, 4, 5, 6, 7,
                                        8, 9, 10, 11, 12, 13, 14, 15);
      acc[j] = __builtin_amdgcn_wmma_f32_16x16x32_bf16(
          false, a, false, b, (short)0, acc[j], false, false);
    }
  }
  const int rbase = m0 + half * 8;            // C layout: lanes 0-15 rows m0.., 16-31 rows m0+8..
#pragma unroll
  for (int j = 0; j < 4; ++j) {
    const int col = n0 + j * 16 + l15;
    const float bv = bias ? bias[col] : 0.0f;
#pragma unroll
    for (int r = 0; r < 8; ++r) {
      int rr = rbase + r;
      if (rr < M) C[(size_t)rr * Nout + col] = acc[j][r] + bv;
    }
  }
}

// transpose+convert fp32 W[K,Nout] -> bf16 Wt[Nout,K]
__global__ void k_wt_convert(const float* __restrict__ W, __bf16* __restrict__ Wt,
                             int K, int Nout)
{
  int idx = blockIdx.x * blockDim.x + threadIdx.x;
  if (idx >= K * Nout) return;
  int k = idx / Nout, n = idx % Nout;
  Wt[(size_t)n * K + k] = (__bf16)W[(size_t)k * Nout + n];
}

// flat fp32 -> bf16 activation convert
__global__ void k_a_convert(const float* __restrict__ A, __bf16* __restrict__ Ab, size_t n)
{
  size_t i = (size_t)blockIdx.x * blockDim.x + threadIdx.x;
  if (i < n) Ab[i] = (__bf16)A[i];
}

// ---------------- elementwise / scatter kernels ----------------
__global__ void k_fill_f(float* p, float v, size_t n) {
  size_t i = (size_t)blockIdx.x * blockDim.x + threadIdx.x;
  if (i < n) p[i] = v;
}
__global__ void k_fill_i(int* p, int v, size_t n) {
  size_t i = (size_t)blockIdx.x * blockDim.x + threadIdx.x;
  if (i < n) p[i] = v;
}

__global__ void k_node_emb(const float* __restrict__ x, const float* __restrict__ Wn,
                           const float* __restrict__ b, float* __restrict__ h)
{
  size_t i = (size_t)blockIdx.x * blockDim.x + threadIdx.x;
  if (i >= (size_t)NN * HIDd) return;
  int n = (int)(i >> 8), j = (int)(i & 255);
  float acc = b[j];
#pragma unroll
  for (int k = 0; k < 10; ++k) acc += x[(size_t)n * 10 + k] * Wn[k * HIDd + j];
  h[i] = acc;
}

__global__ void k_edge_emb(const float* __restrict__ ea, const float* __restrict__ We,
                           const float* __restrict__ b, float* __restrict__ e)
{
  size_t i = (size_t)blockIdx.x * blockDim.x + threadIdx.x;
  if (i >= (size_t)EEg * HIDd) return;
  int n = (int)(i >> 8), j = (int)(i & 255);
  float acc = b[j];
#pragma unroll
  for (int k = 0; k < 4; ++k) acc += ea[(size_t)n * 4 + k] * We[k * HIDd + j];
  e[i] = acc;
}

__global__ void k_deg(const int* __restrict__ dst, float* __restrict__ deg) {
  size_t i = (size_t)blockIdx.x * blockDim.x + threadIdx.x;
  if (i < (size_t)EEg) atomicAdd(&deg[dst[i]], 1.0f);
}
__global__ void k_loop_accum(const float* __restrict__ e, const int* __restrict__ dst,
                             float* __restrict__ loop) {
  size_t t = (size_t)blockIdx.x * blockDim.x + threadIdx.x;
  if (t >= (size_t)EEg * HIDd) return;
  int ed = (int)(t >> 8), j = (int)(t & 255);
  atomicAdd(&loop[(size_t)dst[ed] * HIDd + j], e[t]);
}
__global__ void k_loop_div(float* __restrict__ loop, const float* __restrict__ deg) {
  size_t t = (size_t)blockIdx.x * blockDim.x + threadIdx.x;
  if (t >= (size_t)NN * HIDd) return;
  loop[t] /= fmaxf(deg[t >> 8], 1.0f);
}

// per-node attention scalars: alpha_src/dst[n,h] = <hh[n,h,:], att_{src,dst}[h,:]>
__global__ void k_att_node(const float* __restrict__ hh, const float* __restrict__ as,
                           const float* __restrict__ ad, float* __restrict__ oa,
                           float* __restrict__ ob)
{
  size_t t = (size_t)blockIdx.x * blockDim.x + threadIdx.x;
  if (t >= (size_t)NN * NH) return;
  int n = (int)(t >> 2), hh_ = (int)(t & 3);
  const float* hp = hh + (size_t)n * HIDd + hh_ * HC;
  float a = 0.f, b = 0.f;
  for (int c = 0; c < HC; ++c) { float v = hp[c]; a += v * as[hh_ * HC + c]; b += v * ad[hh_ * HC + c]; }
  oa[t] = a; ob[t] = b;
}

// attW[k,h] = sum_c We[k, h*64+c] * att_edge[h,c]   (collapses ee = e@We against att_edge)
__global__ void k_attw(const float* __restrict__ We, const float* __restrict__ ae,
                       float* __restrict__ attW)
{
  int t = blockIdx.x * blockDim.x + threadIdx.x;
  if (t >= HIDd * NH) return;
  int k = t >> 2, h = t & 3;
  float acc = 0.f;
  for (int c = 0; c < HC; ++c) acc += We[(size_t)k * HIDd + h * HC + c] * ae[h * HC + c];
  attW[t] = acc;
}

__global__ void k_gat_logits(const float* __restrict__ asrc, const float* __restrict__ adst,
                             const float* __restrict__ e_a, const float* __restrict__ attW,
                             const int* __restrict__ src, const int* __restrict__ dst,
                             float* __restrict__ logits, int* __restrict__ segmax)
{
  size_t t = (size_t)blockIdx.x * blockDim.x + threadIdx.x;
  if (t >= (size_t)EAg * NH) return;
  int ed = (int)(t >> 2), h = (int)(t & 3);
  int s, d;
  if (ed >= EEg) { s = d = ed - EEg; } else { s = src[ed]; d = dst[ed]; }
  float a = asrc[(size_t)s * NH + h] + adst[(size_t)d * NH + h];
  const float* ep = e_a + (size_t)ed * HIDd;
  float acc = 0.f;
  for (int k = 0; k < HIDd; ++k) acc += ep[k] * attW[k * NH + h];
  a += acc;
  a = a > 0.f ? a : 0.2f * a;               // leaky relu
  logits[t] = a;
  atomicMax(&segmax[(size_t)d * NH + h], enc_f(a));
}

__global__ void k_expsum(float* __restrict__ logits, const int* __restrict__ dst,
                         const int* __restrict__ segmax, float* __restrict__ segsum,
                         int Etot, int Ereal)
{
  size_t t = (size_t)blockIdx.x * blockDim.x + threadIdx.x;
  if (t >= (size_t)Etot * NH) return;
  int ed = (int)(t >> 2), h = (int)(t & 3);
  int d = (ed >= Ereal) ? ed - Ereal : dst[ed];
  float m = dec_f(segmax[(size_t)d * NH + h]);
  float e = __expf(logits[t] - m);
  logits[t] = e;
  atomicAdd(&segsum[(size_t)d * NH + h], e);
}

__global__ void k_gat_aggr(const float* __restrict__ hh, const float* __restrict__ logits,
                           const float* __restrict__ segsum, const int* __restrict__ src,
                           const int* __restrict__ dst, float* __restrict__ out)
{
  size_t t = (size_t)blockIdx.x * blockDim.x + threadIdx.x;
  if (t >= (size_t)EAg * HIDd) return;
  int ed = (int)(t >> 8), j = (int)(t & 255), h = j >> 6;
  int s, d;
  if (ed >= EEg) { s = d = ed - EEg; } else { s = src[ed]; d = dst[ed]; }
  float al = logits[(size_t)ed * NH + h] / (segsum[(size_t)d * NH + h] + 1e-16f);
  atomicAdd(&out[(size_t)d * HIDd + j], hh[(size_t)s * HIDd + j] * al);
}

__global__ void k_trf_logits(const float* __restrict__ q, const float* __restrict__ kk,
                             const float* __restrict__ ee, const int* __restrict__ src,
                             const int* __restrict__ dst, float* __restrict__ logits,
                             int* __restrict__ segmax)
{
  size_t t = (size_t)blockIdx.x * blockDim.x + threadIdx.x;
  if (t >= (size_t)EEg * NH) return;
  int ed = (int)(t >> 2), h = (int)(t & 3);
  int s = src[ed], d = dst[ed];
  const float* qp = q  + (size_t)d  * HIDd + h * HC;
  const float* kp = kk + (size_t)s  * HIDd + h * HC;
  const float* ep = ee + (size_t)ed * HIDd + h * HC;
  float a = 0.f;
  for (int c = 0; c < HC; ++c) a += qp[c] * (kp[c] + ep[c]);
  a *= 0.125f;                               // 1/sqrt(64)
  logits[t] = a;
  atomicMax(&segmax[(size_t)d * NH + h], enc_f(a));
}

__global__ void k_trf_aggr(const float* __restrict__ v, const float* __restrict__ ee,
                           const float* __restrict__ logits, const float* __restrict__ segsum,
                           const int* __restrict__ src, const int* __restrict__ dst,
                           float* __restrict__ out)
{
  size_t t = (size_t)blockIdx.x * blockDim.x + threadIdx.x;
  if (t >= (size_t)EEg * HIDd) return;
  int ed = (int)(t >> 8), j = (int)(t & 255), h = j >> 6;
  int s = src[ed], d = dst[ed];
  float al = logits[(size_t)ed * NH + h] / (segsum[(size_t)d * NH + h] + 1e-16f);
  atomicAdd(&out[(size_t)d * HIDd + j], (v[(size_t)s * HIDd + j] + ee[t]) * al);
}

__global__ void k_bn_relu_res(const float* __restrict__ in, const float* __restrict__ ab,
                              const float* __restrict__ g, const float* __restrict__ b,
                              const float* __restrict__ mu, const float* __restrict__ var,
                              float* __restrict__ h, size_t total, int D)
{
  size_t i = (size_t)blockIdx.x * blockDim.x + threadIdx.x;
  if (i >= total) return;
  int j = (int)(i % D);
  float x = in[i] + (ab ? ab[j] : 0.0f);
  x = (x - mu[j]) * rsqrtf(var[j] + 1e-5f) * g[j] + b[j];
  h[i] = fmaxf(x, 0.0f) + h[i];
}

__global__ void k_bn_relu(float* __restrict__ f, const float* __restrict__ g,
                          const float* __restrict__ b, const float* __restrict__ mu,
                          const float* __restrict__ var, size_t total, int D)
{
  size_t i = (size_t)blockIdx.x * blockDim.x + threadIdx.x;
  if (i >= total) return;
  int j = (int)(i % D);
  float x = (f[i] - mu[j]) * rsqrtf(var[j] + 1e-5f) * g[j] + b[j];
  f[i] = fmaxf(x, 0.0f);
}

__global__ void k_tanh_ip(float* p, size_t n) {
  size_t i = (size_t)blockIdx.x * blockDim.x + threadIdx.x;
  if (i < n) p[i] = tanhf(p[i]);
}

__global__ void k_gate(const float* __restrict__ t0, const float* __restrict__ A2,
                       const float* __restrict__ a2, float* __restrict__ gate)
{
  size_t n = (size_t)blockIdx.x * blockDim.x + threadIdx.x;
  if (n >= (size_t)NN) return;
  float acc = a2[0];
  const float* tp = t0 + n * HIDd;
  for (int j = 0; j < HIDd; ++j) acc += tp[j] * A2[j];
  gate[n] = 1.0f / (1.0f + __expf(-acc));
}

__global__ void k_cnt(const int* __restrict__ batch, float* __restrict__ cnt) {
  size_t n = (size_t)blockIdx.x * blockDim.x + threadIdx.x;
  if (n < (size_t)NN) atomicAdd(&cnt[batch[n]], 1.0f);
}
__global__ void k_pool_accum(const float* __restrict__ h, const float* __restrict__ gate,
                             const int* __restrict__ batch, float* __restrict__ s)
{
  size_t t = (size_t)blockIdx.x * blockDim.x + threadIdx.x;
  if (t >= (size_t)NN * HIDd) return;
  int n = (int)(t >> 8), j = (int)(t & 255);
  atomicAdd(&s[(size_t)batch[n] * HIDd + j], h[t] * gate[n]);
}
__global__ void k_pooled(const float* __restrict__ s, const float* __restrict__ cnt,
                         float* __restrict__ pooled)
{
  size_t t = (size_t)blockIdx.x * blockDim.x + threadIdx.x;
  if (t >= (size_t)GGr * 512) return;
  int g = (int)(t >> 9), j = (int)(t & 511);
  pooled[t] = (j < HIDd) ? s[(size_t)g * HIDd + j] / fmaxf(cnt[g], 1.0f)
                         : s[(size_t)g * HIDd + (j - HIDd)];
}

__global__ void k_heads(const float* __restrict__ f2,
                        const float* __restrict__ Hf, const float* __restrict__ hf,
                        const float* __restrict__ Hv, const float* __restrict__ hv,
                        const float* __restrict__ Hc, const float* __restrict__ hc,
                        float* __restrict__ out)
{
  size_t g = (size_t)blockIdx.x * blockDim.x + threadIdx.x;
  if (g >= (size_t)GGr) return;
  float a = 0.f, b = 0.f, c = 0.f;
  const float* fp = f2 + g * 128;
  for (int j = 0; j < 128; ++j) { float v = fp[j]; a += v * Hf[j]; b += v * Hv[j]; c += v * Hc[j]; }
  out[g * 3 + 0] = a + hf[0];
  out[g * 3 + 1] = b + hv[0];
  out[g * 3 + 2] = c + hc[0];
}

// ---------------- host orchestration ----------------
extern "C" void kernel_launch(void* const* d_in, const int* in_sizes, int n_in,
                              void* d_out, int out_size, void* d_ws, size_t ws_size,
                              hipStream_t stream)
{
  (void)in_sizes; (void)n_in; (void)out_size; (void)ws_size;
  const float* x     = (const float*)d_in[0];
  const float* eattr = (const float*)d_in[1];
  const int*   eidx  = (const int*)d_in[2];
  const int*   batch = (const int*)d_in[3];
  const int* src = eidx;
  const int* dst = eidx + EEg;
#define P(i) ((const float*)d_in[i])
  // param leaf order (recursive insertion order of setup_inputs params dict):
  // 4:Wn 5:bn_emb 6:We_emb 7:be_emb ; gat_l @ 8+16l {W,We,att_src,att_dst,att_edge,bias}
  // trf_l @ 14+16l {Wq,bq,Wk,bk,Wv,bv,We,be,Wskip,bskip} ; bn_l @ 56+4l {gamma,beta,mean,var}
  // 80:A1 81:a1 82:A2 83:a2 84:M1 85:m1 86:M2 87:m2 88-91:bn_mlp1 92-95:bn_mlp2
  // 96:Hf 97:hf 98:Hv 99:hv 100:Hc 101:hc

  char* wptr = (char*)d_ws;
  auto carve = [&](size_t bytes) -> char* {
    char* p = wptr; wptr += (bytes + 255) & ~(size_t)255; return p;
  };
  float* e_emb  = (float*)carve((size_t)EEg * HIDd * 4);
  float* e_a    = (float*)carve((size_t)EAg * HIDd * 4);   // GAT augmented attrs; reused as TRF ee
  float* h      = (float*)carve((size_t)NN * HIDd * 4);
  float* t0     = (float*)carve((size_t)NN * HIDd * 4);    // hh / q / tanh tmp
  float* t1     = (float*)carve((size_t)NN * HIDd * 4);    // k
  float* t2     = (float*)carve((size_t)NN * HIDd * 4);    // v
  float* outb   = (float*)carve((size_t)NN * HIDd * 4);
  float* asrc   = (float*)carve((size_t)NN * NH * 4);
  float* adst   = (float*)carve((size_t)NN * NH * 4);
  float* logits = (float*)carve((size_t)EAg * NH * 4);
  int*   segmax = (int*)  carve((size_t)NN * NH * 4);
  float* segsum = (float*)carve((size_t)NN * NH * 4);
  float* deg    = (float*)carve((size_t)NN * 4);
  float* attW   = (float*)carve((size_t)HIDd * NH * 4);
  float* gate   = (float*)carve((size_t)NN * 4);
  float* cnt    = (float*)carve((size_t)GGr * 4);
  float* sbuf   = (float*)carve((size_t)GGr * HIDd * 4);
  float* pooled = (float*)carve((size_t)GGr * 512 * 4);
  float* f1     = (float*)carve((size_t)GGr * HIDd * 4);
  float* f2     = (float*)carve((size_t)GGr * 128 * 4);
  __bf16* wt    = (__bf16*)carve((size_t)512 * 256 * 2);   // transposed weight scratch
  __bf16* hb    = (__bf16*)carve((size_t)NN * HIDd * 2);   // bf16 activations (h)
  __bf16* eb    = (__bf16*)carve((size_t)EEg * HIDd * 2);  // bf16 e_emb (TRF layers)
  __bf16* pb    = (__bf16*)carve((size_t)GGr * 512 * 2);   // bf16 pooled
  __bf16* f1b   = (__bf16*)carve((size_t)GGr * HIDd * 2);  // bf16 f1

  const int B = 256;
  auto g1 = [](size_t n) { return dim3((unsigned)((n + 255) / 256)); };
  auto gemm = [&](const __bf16* Ab, const float* W, const float* bias, float* C,
                  int M, int K, int Nout) {
    int tot = K * Nout;
    k_wt_convert<<<(tot + 255) / 256, 256, 0, stream>>>(W, wt, K, Nout);
    dim3 grid((unsigned)((M + 63) / 64), (unsigned)(Nout / 64));
    k_gemm<<<grid, 128, 0, stream>>>(Ab, wt, bias, C, M, K, Nout);
  };
  auto cvt = [&](const float* A, __bf16* Ab, size_t n) {
    k_a_convert<<<g1(n), B, 0, stream>>>(A, Ab, n);
  };

  // embeddings
  k_node_emb<<<g1((size_t)NN * HIDd), B, 0, stream>>>(x, P(4), P(5), h);
  k_edge_emb<<<g1((size_t)EEg * HIDd), B, 0, stream>>>(eattr, P(6), P(7), e_emb);

  // self-loop edge attrs: e_a = [e_emb ; segsum(e_emb)/deg]
  k_fill_f<<<g1(NN), B, 0, stream>>>(deg, 0.f, (size_t)NN);
  k_deg<<<g1(EEg), B, 0, stream>>>(dst, deg);
  hipMemcpyAsync(e_a, e_emb, (size_t)EEg * HIDd * 4, hipMemcpyDeviceToDevice, stream);
  k_fill_f<<<g1((size_t)NN * HIDd), B, 0, stream>>>(e_a + (size_t)EEg * HIDd, 0.f, (size_t)NN * HIDd);
  k_loop_accum<<<g1((size_t)EEg * HIDd), B, 0, stream>>>(e_emb, dst, e_a + (size_t)EEg * HIDd);
  k_loop_div<<<g1((size_t)NN * HIDd), B, 0, stream>>>(e_a + (size_t)EEg * HIDd, deg);

  // e_emb in bf16 once: reused by all 3 TRF edge GEMMs
  cvt(e_emb, eb, (size_t)EEg * HIDd);

  // ---- 3 GAT layers ----
  for (int l = 0; l < 3; ++l) {
    int gb = 8 + 16 * l, bnb = 56 + 4 * l;
    cvt(h, hb, (size_t)NN * HIDd);                                       // h -> bf16 once/layer
    gemm(hb, P(gb + 0), nullptr, t0, NN, HIDd, HIDd);                    // hh = h@W  (WMMA)
    k_att_node<<<g1((size_t)NN * NH), B, 0, stream>>>(t0, P(gb + 2), P(gb + 3), asrc, adst);
    k_attw<<<g1((size_t)HIDd * NH), B, 0, stream>>>(P(gb + 1), P(gb + 4), attW);
    k_fill_i<<<g1((size_t)NN * NH), B, 0, stream>>>(segmax, (-2147483647 - 1), (size_t)NN * NH);
    k_fill_f<<<g1((size_t)NN * NH), B, 0, stream>>>(segsum, 0.f, (size_t)NN * NH);
    k_fill_f<<<g1((size_t)NN * HIDd), B, 0, stream>>>(outb, 0.f, (size_t)NN * HIDd);
    k_gat_logits<<<g1((size_t)EAg * NH), B, 0, stream>>>(asrc, adst, e_a, attW, src, dst, logits, segmax);
    k_expsum<<<g1((size_t)EAg * NH), B, 0, stream>>>(logits, dst, segmax, segsum, EAg, EEg);
    k_gat_aggr<<<g1((size_t)EAg * HIDd), B, 0, stream>>>(t0, logits, segsum, src, dst, outb);
    k_bn_relu_res<<<g1((size_t)NN * HIDd), B, 0, stream>>>(
        outb, P(gb + 5), P(bnb), P(bnb + 1), P(bnb + 2), P(bnb + 3), h, (size_t)NN * HIDd, HIDd);
  }

  // ---- 3 TransformerConv layers ----
  for (int l = 0; l < 3; ++l) {
    int tb = 14 + 16 * l, bnb = 56 + 4 * (3 + l);
    cvt(h, hb, (size_t)NN * HIDd);                                       // shared by q,k,v,skip
    gemm(hb, P(tb + 0), P(tb + 1), t0, NN, HIDd, HIDd);                  // q
    gemm(hb, P(tb + 2), P(tb + 3), t1, NN, HIDd, HIDd);                  // k
    gemm(hb, P(tb + 4), P(tb + 5), t2, NN, HIDd, HIDd);                  // v
    gemm(eb, P(tb + 6), P(tb + 7), e_a, EEg, HIDd, HIDd);                // ee (aliases e_a)
    gemm(hb, P(tb + 8), P(tb + 9), outb, NN, HIDd, HIDd);                // skip init
    k_fill_i<<<g1((size_t)NN * NH), B, 0, stream>>>(segmax, (-2147483647 - 1), (size_t)NN * NH);
    k_fill_f<<<g1((size_t)NN * NH), B, 0, stream>>>(segsum, 0.f, (size_t)NN * NH);
    k_trf_logits<<<g1((size_t)EEg * NH), B, 0, stream>>>(t0, t1, e_a, src, dst, logits, segmax);
    k_expsum<<<g1((size_t)EEg * NH), B, 0, stream>>>(logits, dst, segmax, segsum, EEg, EEg);
    k_trf_aggr<<<g1((size_t)EEg * HIDd), B, 0, stream>>>(t2, e_a, logits, segsum, src, dst, outb);
    k_bn_relu_res<<<g1((size_t)NN * HIDd), B, 0, stream>>>(
        outb, nullptr, P(bnb), P(bnb + 1), P(bnb + 2), P(bnb + 3), h, (size_t)NN * HIDd, HIDd);
  }

  // ---- gated pooling ----
  cvt(h, hb, (size_t)NN * HIDd);
  gemm(hb, P(80), P(81), t0, NN, HIDd, HIDd);                            // h@A1 + a1
  k_tanh_ip<<<g1((size_t)NN * HIDd), B, 0, stream>>>(t0, (size_t)NN * HIDd);
  k_gate<<<g1(NN), B, 0, stream>>>(t0, P(82), P(83), gate);
  k_fill_f<<<g1(GGr), B, 0, stream>>>(cnt, 0.f, (size_t)GGr);
  k_fill_f<<<g1((size_t)GGr * HIDd), B, 0, stream>>>(sbuf, 0.f, (size_t)GGr * HIDd);
  k_cnt<<<g1(NN), B, 0, stream>>>(batch, cnt);
  k_pool_accum<<<g1((size_t)NN * HIDd), B, 0, stream>>>(h, gate, batch, sbuf);
  k_pooled<<<g1((size_t)GGr * 512), B, 0, stream>>>(sbuf, cnt, pooled);

  // ---- MLP head ----
  cvt(pooled, pb, (size_t)GGr * 512);
  gemm(pb, P(84), P(85), f1, GGr, 512, HIDd);                            // M1
  k_bn_relu<<<g1((size_t)GGr * HIDd), B, 0, stream>>>(f1, P(88), P(89), P(90), P(91),
                                                      (size_t)GGr * HIDd, HIDd);
  cvt(f1, f1b, (size_t)GGr * HIDd);
  gemm(f1b, P(86), P(87), f2, GGr, HIDd, 128);                           // M2
  k_bn_relu<<<g1((size_t)GGr * 128), B, 0, stream>>>(f2, P(92), P(93), P(94), P(95),
                                                     (size_t)GGr * 128, 128);
  k_heads<<<g1(GGr), B, 0, stream>>>(f2, P(96), P(97), P(98), P(99), P(100), P(101),
                                     (float*)d_out);
#undef P
}
